// ChameleonAttention_32246614458379
// MI455X (gfx1250) — matrix operand order repaired and checked by reading, not compiled, not run on hardware
//
#include <hip/hip_runtime.h>
#include <hip/hip_bf16.h>

// ---------------------------------------------------------------------------
// ChameleonAttention for MI455X (gfx1250):
//   bf16 WMMA GEMMs (async-to-LDS double buffered) + TDM-fed flash attention
// ---------------------------------------------------------------------------

typedef __attribute__((ext_vector_type(16))) __bf16 v16bf;
typedef __attribute__((ext_vector_type(8)))  float  v8f;
typedef __attribute__((ext_vector_type(4)))  int    v4i_;
typedef __attribute__((ext_vector_type(8)))  int    v8i_;

#define T_    2048
#define H_    32
#define KVH_  32
#define D_    128
#define HID_  4096
#define QKVN_ 12288   // (H + 2*KVH) * D

union FragBF { v16bf v; uint4 u[2]; };

// A-matrix 16x32 bf16 fragment from row-major LDS tile (stride ld halves).
// ISA 7.12.2: lanes 0-15 hold rows 0-15 with K 0..7 (v0-3) and 16..23 (v4-7);
// lanes 16-31 hold same rows with K 8..15 and 24..31.
__device__ inline v16bf lds_load_a16x32(const __bf16* base, int row0, int k0, int ld) {
  const int lane = threadIdx.x & 31;
  const int r  = row0 + (lane & 15);
  const int kh = (lane >> 4) << 3;               // 0 or 8
  const __bf16* p = base + r * ld + k0 + kh;
  FragBF f;
  f.u[0] = *reinterpret_cast<const uint4*>(p);       // K = kh .. kh+7
  f.u[1] = *reinterpret_cast<const uint4*>(p + 16);  // K = kh+16 .. kh+23
  return f.v;
}

// B-matrix 32x16 bf16 fragment; LDS tile stored as [n][k] row-major (k contig).
// Lane L holds column N = L%16; halves 0-7 hold K = (L/16)*8 + j, halves 8-15 hold K = 16 + (L/16)*8 + (j-8).
__device__ inline v16bf lds_load_b32x16(const __bf16* base, int n0, int k0, int ld) {
  const int lane = threadIdx.x & 31;
  const int col = n0 + (lane & 15);
  const int ko  = (lane >> 4) << 3;              // 0 or 8
  const __bf16* p = base + col * ld + k0 + ko;
  FragBF f;
  f.u[0] = *reinterpret_cast<const uint4*>(p);
  f.u[1] = *reinterpret_cast<const uint4*>(p + 16);
  return f.v;
}

// Async 16B copy global -> LDS (GLOBAL_LOAD_ASYNC_TO_LDS_B128, ASYNCcnt).
__device__ inline void async_copy_b128(unsigned lds_byte_addr, const __bf16* g) {
  asm volatile("global_load_async_to_lds_b128 %0, %1, off"
               :: "v"(lds_byte_addr), "v"((unsigned long long)(uintptr_t)g)
               : "memory");
}
__device__ inline void wait_async_all() {
  asm volatile("s_wait_asynccnt 0x0" ::: "memory");
}

// TDM 2D tile load: rows x 128 bf16 each, row stride in elements,
// into LDS with padded stride 136 halves (pad_interval=64 DW, pad_amount=4 DW).
__device__ inline void tdm_load_tile_128(unsigned lds_byte_addr, const void* gaddr,
                                         int rows, long row_stride_elts) {
  const unsigned long long ga = (unsigned long long)(uintptr_t)gaddr;
  v4i_ g0;
  g0[0] = 1;                                            // count=1, user D#
  g0[1] = (int)lds_byte_addr;                           // lds_addr
  g0[2] = (int)(ga & 0xffffffffull);                    // global_addr[31:0]
  g0[3] = (int)(((ga >> 32) & 0x1ffffffull) | (2u << 30)); // addr[56:32], type=2
  v8i_ g1;
  g1[0] = (1 << 16)                                     // data_size = 2 bytes
        | (1 << 20)                                     // pad_enable
        | (5 << 22)                                     // pad_interval: 64 DWORDs
        | (3 << 25);                                    // pad_amount: 4 DWORDs
  g1[1] = 128 << 16;                                    // tensor_dim0 = 128 (lo16)
  g1[2] = (rows & 0xffff) << 16;                        // tensor_dim1 (lo16)
  g1[3] = 128 << 16;                                    // tile_dim0 = 128
  g1[4] = rows & 0xffff;                                // tile_dim1; tile_dim2=0
  g1[5] = (int)row_stride_elts;                         // tensor_dim0_stride lo32
  g1[6] = 0;                                            // stride hi / dim1_stride
  g1[7] = 0;
  asm volatile("tensor_load_to_lds %0, %1" :: "s"(g0), "s"(g1) : "memory");
}

// ---------------------------------------------------------------------------
// Kernel 1: fp32 -> bf16 cast (vectorized), n4 = count/4
// ---------------------------------------------------------------------------
__global__ void __launch_bounds__(256)
cast_f32_bf16(const float* __restrict__ src, __bf16* __restrict__ dst, long n4) {
  long i = (long)blockIdx.x * blockDim.x + threadIdx.x;
  const long stride = (long)gridDim.x * blockDim.x;
  const float4* s4 = reinterpret_cast<const float4*>(src);
  ushort4* d4 = reinterpret_cast<ushort4*>(dst);
  for (; i < n4; i += stride) {
    float4 v = s4[i];
    union { __bf16 h[4]; ushort4 u; } r;
    r.h[0] = (__bf16)v.x; r.h[1] = (__bf16)v.y;
    r.h[2] = (__bf16)v.z; r.h[3] = (__bf16)v.w;
    d4[i] = r.u;
  }
}

// ---------------------------------------------------------------------------
// Kernel 2/5: C[M,N] (fp32) = A[M,K] (bf16 rm) * B[N,K]^T (bf16 rm)
// 128x128 tile, K-block 64, double-buffered LDS fed by async-to-LDS copies.
// 8 waves in 4(M) x 2(N); each wave: 2x4 WMMA 16x16x32 bf16 tiles.
// NOTE: no pointer arrays built from the extern-shared symbol (ld.lld cannot
// relocate addrspacecast constants) — buffer addresses are computed scalars.
// ---------------------------------------------------------------------------
__global__ void __launch_bounds__(256)
gemm_bf16_nt(const __bf16* __restrict__ A, const __bf16* __restrict__ B,
             float* __restrict__ C, int M, int N, int K) {
  extern __shared__ __bf16 smem[];
  constexpr int LD = 72;                         // 64 + 8 pad halves (144B rows)
  constexpr int TILE = 128 * LD;                 // halves per tile buffer

  const int tid  = threadIdx.x;
  const int wave = tid >> 5;
  const int lane = tid & 31;
  const int lo = lane & 15, hi = lane >> 4;
  const int m0 = blockIdx.y * 128;
  const int n0 = blockIdx.x * 128;
  const int wm = (wave & 3) * 32;
  const int wn = (wave >> 2) * 64;

  const int lrow = tid >> 1;                     // 0..127
  const int lcol = (tid & 1) * 32;               // 0 or 32

  v8f acc[2][4];
  v8f vzero = {};
#pragma unroll
  for (int i = 0; i < 2; ++i)
#pragma unroll
    for (int j = 0; j < 4; ++j) acc[i][j] = vzero;

  const __bf16* gA = A + (size_t)(m0 + lrow) * K + lcol;
  const __bf16* gB = B + (size_t)(n0 + lrow) * K + lcol;
  // Per-thread LDS byte address of this thread's slice in buffer 0 of A.
  const unsigned ldsThr =
      (unsigned)(uintptr_t)smem + (unsigned)(lrow * LD + lcol) * 2u;

  // Each thread async-copies 32 halves (4 x b128) of A and of B per tile.
  auto issue_tile = [&](int buf, int kb) {
    const unsigned la = ldsThr + (unsigned)(buf * (2 * TILE) * 2);
    const unsigned lb = la + (unsigned)(TILE * 2);
#pragma unroll
    for (int i = 0; i < 4; ++i) {
      async_copy_b128(la + i * 16, gA + kb + i * 8);
      async_copy_b128(lb + i * 16, gB + kb + i * 8);
    }
  };

  issue_tile(0, 0);
  wait_async_all();
  __syncthreads();

  const int nkb = K >> 6;
  for (int ib = 0; ib < nkb; ++ib) {
    const int cur = ib & 1;
    if (ib + 1 < nkb) issue_tile(cur ^ 1, (ib + 1) << 6);  // DMA overlaps WMMA

    const __bf16* tA = smem + cur * (2 * TILE);
    const __bf16* tB = tA + TILE;
#pragma unroll
    for (int kc = 0; kc < 2; ++kc) {
      v16bf af[2];
#pragma unroll
      for (int mt = 0; mt < 2; ++mt) af[mt] = lds_load_a16x32(tA, wm + mt * 16, kc * 32, LD);
#pragma unroll
      for (int nt = 0; nt < 4; ++nt) {
        v16bf bf = lds_load_b32x16(tB, wn + nt * 16, kc * 32, LD);
#pragma unroll
        for (int mt = 0; mt < 2; ++mt)
          acc[mt][nt] = __builtin_amdgcn_wmma_f32_16x16x32_bf16(
              false, af[mt], false, bf, (short)0, acc[mt][nt], false, false);
      }
    }
    wait_async_all();                            // my async writes to nxt done
    __syncthreads();                             // all waves' writes visible
  }

  // C/D layout: lane L, vgpr j -> row = 8*(L/16)+j, col = L%16
#pragma unroll
  for (int mt = 0; mt < 2; ++mt)
#pragma unroll
    for (int nt = 0; nt < 4; ++nt)
#pragma unroll
      for (int j = 0; j < 8; ++j) {
        int row = m0 + wm + mt * 16 + hi * 8 + j;
        int col = n0 + wn + nt * 16 + lo;
        C[(size_t)row * N + col] = acc[mt][nt][j];
      }
}

// ---------------------------------------------------------------------------
// Kernel 3: per-head LayerNorm + RoPE on q/k, cast v. One wave per (token,head).
// ---------------------------------------------------------------------------
__global__ void __launch_bounds__(256)
lnrope_kernel(const int* __restrict__ positions, const float* __restrict__ qkv,
              const float* __restrict__ qw, const float* __restrict__ qb,
              const float* __restrict__ kw, const float* __restrict__ kb,
              __bf16* __restrict__ qo, __bf16* __restrict__ ko,
              __bf16* __restrict__ vo) {
  const int t = blockIdx.x;
  const int tid = threadIdx.x, wave = tid >> 5, lane = tid & 31;
  const float pos = (float)positions[t];
  const float* row = qkv + (size_t)t * QKVN_;
  const float LN_INV = -0.14391156f;             // -ln(10000)/64

  for (int slot = wave; slot < H_ + KVH_; slot += 8) {
    const bool isQ = slot < H_;
    const int h = isQ ? slot : slot - H_;
    const float* x = row + (isQ ? h * D_ : HID_ + h * D_);
    float v0 = x[lane], v1 = x[lane + 32], v2 = x[lane + 64], v3 = x[lane + 96];
    float s = v0 + v1 + v2 + v3;
#pragma unroll
    for (int d = 1; d < 32; d <<= 1) s += __shfl_xor(s, d, 32);
    float mu = s * (1.0f / 128.0f);
    float e0 = v0 - mu, e1 = v1 - mu, e2 = v2 - mu, e3 = v3 - mu;
    float sq = e0 * e0 + e1 * e1 + e2 * e2 + e3 * e3;
#pragma unroll
    for (int d = 1; d < 32; d <<= 1) sq += __shfl_xor(sq, d, 32);
    float rstd = rsqrtf(sq * (1.0f / 128.0f) + 1e-5f);
    const float* w = (isQ ? qw : kw) + h * D_;
    const float* b = (isQ ? qb : kb) + h * D_;
    float n0 = e0 * rstd * w[lane]      + b[lane];
    float n1 = e1 * rstd * w[lane + 32] + b[lane + 32];
    float n2 = e2 * rstd * w[lane + 64] + b[lane + 64];
    float n3 = e3 * rstd * w[lane + 96] + b[lane + 96];
    // RoPE: pairs (d, d+64); freq index i = d for d in [0,64)
    float a0 = pos * __expf((float)lane * LN_INV);
    float a1 = pos * __expf((float)(lane + 32) * LN_INV);
    float s0, c0, s1, c1;
    __sincosf(a0, &s0, &c0);
    __sincosf(a1, &s1, &c1);
    __bf16* dst = (isQ ? qo : ko) + ((size_t)t * 32 + h) * D_;
    dst[lane]      = (__bf16)(n0 * c0 - n2 * s0);
    dst[lane + 64] = (__bf16)(n2 * c0 + n0 * s0);
    dst[lane + 32] = (__bf16)(n1 * c1 - n3 * s1);
    dst[lane + 96] = (__bf16)(n3 * c1 + n1 * s1);
  }
  for (int i = tid; i < KVH_ * D_; i += 256)
    vo[(size_t)t * (KVH_ * D_) + i] = (__bf16)row[2 * HID_ + i];
}

// ---------------------------------------------------------------------------
// Kernel 4: causal flash attention. Block = (q-tile of 128 rows, head).
// K/V tiles arrive via the Tensor Data Mover (tensor_load_to_lds, TENSORcnt);
// 8 waves; wave w owns q rows [16w,16w+16). S and PV both bf16 WMMA.
// Dynamic LDS: sQ, sK, sV, sVt, sP — each 128 x 136 bf16 (170 KB).
// ---------------------------------------------------------------------------
__global__ void __launch_bounds__(256)
flash_attn(const __bf16* __restrict__ Q, const __bf16* __restrict__ Km,
           const __bf16* __restrict__ V, __bf16* __restrict__ O) {
  extern __shared__ __bf16 smem[];
  constexpr int LD = 136;                        // 128 + 8 pad
  __bf16* sQ  = smem;
  __bf16* sK  = sQ  + 128 * LD;
  __bf16* sV  = sK  + 128 * LD;                  // [tok][d] (TDM destination)
  __bf16* sVt = sV  + 128 * LD;                  // [d][tok]
  __bf16* sP  = sVt + 128 * LD;

  const int qt = blockIdx.x, h = blockIdx.y;
  const int q0 = qt * 128;
  const int tid = threadIdx.x, wave = tid >> 5, lane = tid & 31;
  const int lo = lane & 15, hi = lane >> 4;
  const int wrow = wave * 16;
  const float scale = 0.08838834764831845f;      // 1/sqrt(128)

  // Load the Q tile once (plain b128 vector loads).
  {
    const int r = tid >> 1, cb = (tid & 1) * 64;
    const uint4* g = reinterpret_cast<const uint4*>(Q + ((size_t)(q0 + r) * H_ + h) * D_ + cb);
    uint4* d = reinterpret_cast<uint4*>(&sQ[r * LD + cb]);
#pragma unroll
    for (int i = 0; i < 8; ++i) d[i] = g[i];
  }

  float m_[8], l_[8];
  v8f o_acc[8];
  v8f vzero = {};
#pragma unroll
  for (int j = 0; j < 8; ++j) { m_[j] = -1e30f; l_[j] = 0.0f; }
#pragma unroll
  for (int nt = 0; nt < 8; ++nt) o_acc[nt] = vzero;

  for (int kt = 0; kt <= qt; ++kt) {
    const int k0 = kt * 128;
    __syncthreads();                             // prior reads of sK/sV/sVt done

    if (wave == 0) {                             // TDM: DMA K and V tiles to LDS
      tdm_load_tile_128((unsigned)(uintptr_t)sK,
                        Km + ((size_t)k0 * KVH_ + h) * D_, 128, (long)KVH_ * D_);
      tdm_load_tile_128((unsigned)(uintptr_t)sV,
                        V + ((size_t)k0 * KVH_ + h) * D_, 128, (long)KVH_ * D_);
      __builtin_amdgcn_s_wait_tensorcnt(0);
    }
    __syncthreads();                             // publish sK / sV

    // Transpose V within LDS: sV[tok][d] -> sVt[d][tok]
    {
      const int r = tid >> 1, cb = (tid & 1) * 64;
      uint4 rv[8];
      const uint4* sv = reinterpret_cast<const uint4*>(&sV[r * LD + cb]);
#pragma unroll
      for (int i = 0; i < 8; ++i) rv[i] = sv[i];
      const __bf16* hv = reinterpret_cast<const __bf16*>(rv);
#pragma unroll
      for (int i = 0; i < 64; ++i) sVt[(cb + i) * LD + r] = hv[i];
    }
    __syncthreads();                             // publish sVt

    // S = Q K^T  (per wave: 16 rows x 128 cols)
    v8f s_acc[8];
#pragma unroll
    for (int nt = 0; nt < 8; ++nt) s_acc[nt] = vzero;
#pragma unroll
    for (int kc = 0; kc < 4; ++kc) {
      v16bf a = lds_load_a16x32(sQ, wrow, kc * 32, LD);
#pragma unroll
      for (int nt = 0; nt < 8; ++nt) {
        v16bf b = lds_load_b32x16(sK, nt * 16, kc * 32, LD);
        s_acc[nt] = __builtin_amdgcn_wmma_f32_16x16x32_bf16(
            false, a, false, b, (short)0, s_acc[nt], false, false);
      }
    }

    // Scale + causal mask (only diagonal tile needs masking).
#pragma unroll
    for (int nt = 0; nt < 8; ++nt)
#pragma unroll
      for (int j = 0; j < 8; ++j) {
        float v = s_acc[nt][j] * scale;
        if (kt == qt) {
          int col  = k0 + nt * 16 + lo;
          int rrow = q0 + wrow + hi * 8 + j;
          if (col > rrow) v = -1e30f;
        }
        s_acc[nt][j] = v;
      }

    // Online softmax (row = 8*hi + j within wave tile; 16-lane reductions).
#pragma unroll
    for (int j = 0; j < 8; ++j) {
      float mx = -1e30f;
#pragma unroll
      for (int nt = 0; nt < 8; ++nt) mx = fmaxf(mx, s_acc[nt][j]);
#pragma unroll
      for (int d = 1; d < 16; d <<= 1) mx = fmaxf(mx, __shfl_xor(mx, d, 32));
      float mnew  = fmaxf(m_[j], mx);
      float alpha = __expf(m_[j] - mnew);
      m_[j] = mnew;
      float rs = 0.0f;
#pragma unroll
      for (int nt = 0; nt < 8; ++nt) {
        float p = __expf(s_acc[nt][j] - mnew);
        s_acc[nt][j] = p;
        rs += p;
      }
#pragma unroll
      for (int d = 1; d < 16; d <<= 1) rs += __shfl_xor(rs, d, 32);
      l_[j] = l_[j] * alpha + rs;
#pragma unroll
      for (int nt = 0; nt < 8; ++nt) o_acc[nt][j] *= alpha;
    }

    // P (bf16) to wave-private LDS rows, then O += P * V.
#pragma unroll
    for (int nt = 0; nt < 8; ++nt)
#pragma unroll
      for (int j = 0; j < 8; ++j)
        sP[(wrow + hi * 8 + j) * LD + nt * 16 + lo] = (__bf16)s_acc[nt][j];

#pragma unroll
    for (int kc = 0; kc < 4; ++kc) {
      v16bf a = lds_load_a16x32(sP, wrow, kc * 32, LD);
#pragma unroll
      for (int nt = 0; nt < 8; ++nt) {
        v16bf b = lds_load_b32x16(sVt, nt * 16, kc * 32, LD);  // [d][tok]: k contig
        o_acc[nt] = __builtin_amdgcn_wmma_f32_16x16x32_bf16(
            false, a, false, b, (short)0, o_acc[nt], false, false);
      }
    }
  }

  // Epilogue: O /= l, write bf16 [T, H*D]
#pragma unroll
  for (int nt = 0; nt < 8; ++nt)
#pragma unroll
    for (int j = 0; j < 8; ++j) {
      int row = q0 + wrow + hi * 8 + j;
      int col = h * D_ + nt * 16 + lo;
      O[(size_t)row * (H_ * D_) + col] = (__bf16)(o_acc[nt][j] / l_[j]);
    }
}

// ---------------------------------------------------------------------------
// Host launcher
// ---------------------------------------------------------------------------
extern "C" void kernel_launch(void* const* d_in, const int* in_sizes, int n_in,
                              void* d_out, int out_size, void* d_ws, size_t ws_size,
                              hipStream_t stream) {
  const int*   positions = (const int*)d_in[0];
  const float* hidden    = (const float*)d_in[1];
  const float* w_qkv     = (const float*)d_in[2];
  const float* w_o       = (const float*)d_in[3];
  const float* q_norm_w  = (const float*)d_in[4];
  const float* q_norm_b  = (const float*)d_in[5];
  const float* k_norm_w  = (const float*)d_in[6];
  const float* k_norm_b  = (const float*)d_in[7];
  float* out = (float*)d_out;

  char* ws = (char*)d_ws;
  size_t off = 0;
  __bf16* hid_bf  = (__bf16*)(ws + off); off += (size_t)T_ * HID_ * 2;        // 16 MB
  __bf16* wqkv_bf = (__bf16*)(ws + off); off += (size_t)QKVN_ * HID_ * 2;     // 96 MB
  __bf16* wo_bf   = (__bf16*)(ws + off); off += (size_t)HID_ * H_ * D_ * 2;   // 32 MB
  float*  qkv_f   = (float*) (ws + off); off += (size_t)T_ * QKVN_ * 4;       // 96 MB
  __bf16* q_bf    = (__bf16*)(ws + off); off += (size_t)T_ * H_ * D_ * 2;
  __bf16* k_bf    = (__bf16*)(ws + off); off += (size_t)T_ * KVH_ * D_ * 2;
  __bf16* v_bf    = (__bf16*)(ws + off); off += (size_t)T_ * KVH_ * D_ * 2;
  __bf16* attn_bf = (__bf16*)(ws + off); off += (size_t)T_ * H_ * D_ * 2;

  // 1) bf16 casts
  cast_f32_bf16<<<2048, 256, 0, stream>>>(hidden, hid_bf, (long)T_ * HID_ / 4);
  cast_f32_bf16<<<8192, 256, 0, stream>>>(w_qkv, wqkv_bf, (long)QKVN_ * HID_ / 4);
  cast_f32_bf16<<<4096, 256, 0, stream>>>(w_o, wo_bf, (long)HID_ * H_ * D_ / 4);

  // 2) QKV = hidden @ w_qkv^T   (M=2048, N=12288, K=4096)
  const size_t gemm_lds = 4u * 128u * 72u * sizeof(__bf16);     // 73728 B
  gemm_bf16_nt<<<dim3(QKVN_ / 128, T_ / 128), 256, gemm_lds, stream>>>(
      hid_bf, wqkv_bf, qkv_f, T_, QKVN_, HID_);

  // 3) per-head LN + RoPE (+ v cast)
  lnrope_kernel<<<T_, 256, 0, stream>>>(positions, qkv_f, q_norm_w, q_norm_b,
                                        k_norm_w, k_norm_b, q_bf, k_bf, v_bf);

  // 4) causal flash attention (dynamic LDS = 5 tiles of 128x136 bf16)
  const size_t flash_lds = 5u * 128u * 136u * sizeof(__bf16);   // 174080 B
  flash_attn<<<dim3(T_ / 128, H_), 256, flash_lds, stream>>>(q_bf, k_bf, v_bf, attn_bf);

  // 5) out = attn @ w_o^T   (M=2048, N=4096, K=4096)
  gemm_bf16_nt<<<dim3(HID_ / 128, T_ / 128), 256, gemm_lds, stream>>>(
      attn_bf, wo_bf, out, T_, HID_, HID_);
}
